// Qwen3Attention_1486058684883
// MI455X (gfx1250) — matrix operand order repaired and checked by reading, not compiled, not run on hardware
//
#include <hip/hip_runtime.h>
#include <hip/hip_bf16.h>
#include <math.h>

typedef __attribute__((ext_vector_type(16))) __bf16 v16bf;
typedef __attribute__((ext_vector_type(8)))  __bf16 v8bf;
typedef __attribute__((ext_vector_type(8)))  float   v8f;
typedef unsigned short u16;

static constexpr int S_LEN = 2048;
static constexpr int DM    = 4096;
static constexpr int NH    = 32;
static constexpr int NKV   = 8;
static constexpr int HD    = 128;
static constexpr int NQKV  = NH*HD + 2*NKV*HD;   // 6144
static constexpr float ATTN_SCALE = 0.08838834764831845f; // 1/sqrt(128)

__device__ __forceinline__ u16 f32_to_bf16(float f) {
  unsigned u = __float_as_uint(f);
  u += 0x7FFFu + ((u >> 16) & 1u);   // round-to-nearest-even
  return (u16)(u >> 16);
}
__device__ __forceinline__ v8f v8f_zero() {
  v8f z;
#pragma unroll
  for (int i = 0; i < 8; ++i) z[i] = 0.0f;
  return z;
}
// Assemble a 16-bit-element A fragment: lane holds row M=la; K-chunks
// [k0+aoff, +8) and [k0+aoff+8, +8) (contiguous 16 K per lane half).
__device__ __forceinline__ v16bf load_afrag(const u16* p /* row base + aoff */,
                                            int k0) {
  v8bf c0 = *reinterpret_cast<const v8bf*>(p + k0);
  v8bf c1 = *reinterpret_cast<const v8bf*>(p + k0 + 8);
  v16bf f;
#pragma unroll
  for (int i = 0; i < 8; ++i) { f[i] = c0[i]; f[i+8] = c1[i]; }
  return f;
}

// -------------------------------------------------------------------------
// f32 -> bf16 conversion (n multiple of 256)
// -------------------------------------------------------------------------
__global__ __launch_bounds__(256) void cvt_bf16_kernel(const float* __restrict__ in,
                                                       u16* __restrict__ out, int n) {
  int i = blockIdx.x * 256 + threadIdx.x;
  if (i < n) out[i] = f32_to_bf16(in[i]);
}

// -------------------------------------------------------------------------
// GEMM helpers: load one k-chunk of fragments / issue the 8 WMMAs.
// -------------------------------------------------------------------------
__device__ __forceinline__ void load_frags(const u16* arow0, const u16* arow1,
                                           const u16* const wrow[4], int k0,
                                           v16bf& a0, v16bf& a1, v16bf b[4]) {
  a0 = load_afrag(arow0, k0);
  a1 = load_afrag(arow1, k0);
#pragma unroll
  for (int t = 0; t < 4; ++t)
    b[t] = *reinterpret_cast<const v16bf*>(wrow[t] + k0);
}
__device__ __forceinline__ void wmma8(const v16bf& a0, const v16bf& a1,
                                      const v16bf b[4], v8f acc[2][4]) {
#pragma unroll
  for (int t = 0; t < 4; ++t)
    acc[0][t] = __builtin_amdgcn_wmma_f32_16x16x32_bf16(false, a0, false, b[t],
                                                        (short)0, acc[0][t], false, false);
#pragma unroll
  for (int t = 0; t < 4; ++t)
    acc[1][t] = __builtin_amdgcn_wmma_f32_16x16x32_bf16(false, a1, false, b[t],
                                                        (short)0, acc[1][t], false, false);
}

// -------------------------------------------------------------------------
// C[M,N](f32) = A[M,K](bf16 row-major) * W[N,K]^T (bf16 row-major).
// Block 256 thr = 8 waves in 2(M) x 4(N); wave tile 32(M) x 64(N);
// block tile 64(M) x 256(N). Ping-pong software pipeline (unroll-by-2,
// K multiple of 64): loads of the next chunk are issued before the WMMAs
// of the current one, and buffers swap roles so no register copies are
// needed for rotation.
// -------------------------------------------------------------------------
__global__ __launch_bounds__(256) void gemm_bf16_wmma(const u16* __restrict__ A,
                                                      const u16* __restrict__ W,
                                                      float* __restrict__ C,
                                                      int M, int N, int K) {
  const int lane = threadIdx.x & 31;
  const int wave = threadIdx.x >> 5;
  const int wm   = wave >> 2;            // 0..1
  const int wn   = wave & 3;             // 0..3
  const int la   = lane & 15;
  const int hi   = lane >> 4;
  const int m0   = blockIdx.y * 64 + wm * 32;
  const int n0   = blockIdx.x * 256 + wn * 64;
  const int aoff = hi ? 16 : 0;

  v8f acc[2][4];
#pragma unroll
  for (int mi = 0; mi < 2; ++mi)
#pragma unroll
    for (int t = 0; t < 4; ++t) acc[mi][t] = v8f_zero();

  const u16* arow0 = A + (size_t)(m0 + la) * K + aoff;
  const u16* arow1 = A + (size_t)(m0 + 16 + la) * K + aoff;
  const u16* wrow[4];
#pragma unroll
  for (int t = 0; t < 4; ++t)
    wrow[t] = W + (size_t)(n0 + t*16 + la) * K + hi*16;

  v16bf a0A, a1A, bA[4];   // ping buffer
  v16bf a0B, a1B, bB[4];   // pong buffer
  load_frags(arow0, arow1, wrow, 0, a0A, a1A, bA);

  int k0 = 0;
  for (; k0 + 64 < K; k0 += 64) {
    load_frags(arow0, arow1, wrow, k0 + 32, a0B, a1B, bB);
    wmma8(a0A, a1A, bA, acc);
    load_frags(arow0, arow1, wrow, k0 + 64, a0A, a1A, bA);
    wmma8(a0B, a1B, bB, acc);
  }
  // tail: chunks k0 (in A) and k0+32
  load_frags(arow0, arow1, wrow, k0 + 32, a0B, a1B, bB);
  wmma8(a0A, a1A, bA, acc);
  wmma8(a0B, a1B, bB, acc);

  // C/D layout: VGPR r -> row r (lanes 0-15) / r+8 (lanes 16-31), col = la
#pragma unroll
  for (int mi = 0; mi < 2; ++mi)
#pragma unroll
    for (int t = 0; t < 4; ++t)
#pragma unroll
      for (int r = 0; r < 8; ++r)
        C[(size_t)(m0 + mi*16 + r + hi*8) * N + (n0 + t*16 + la)] = acc[mi][t][r];
}

// -------------------------------------------------------------------------
// Fused per-head RMSNorm (fp32) + RoPE + pack to bf16.
// grid (S, 48): y in [0,32) = q heads, [32,40) = k heads, [40,48) = v heads.
// q -> qb[H][S][HD], k -> kb[KV][S][HD], v -> vT[KV][HD][S].
// -------------------------------------------------------------------------
__global__ __launch_bounds__(128) void normropepack_kernel(
    const float* __restrict__ qkv, const float* __restrict__ cosb,
    const float* __restrict__ sinb, const float* __restrict__ qw,
    const float* __restrict__ kw, u16* __restrict__ qb, u16* __restrict__ kb,
    u16* __restrict__ vT) {
  const int s = blockIdx.x;
  const int hidx = blockIdx.y;
  const int d = threadIdx.x;
  const float* base = qkv + (size_t)s * NQKV;
  if (hidx >= 40) {  // V: plain convert, transposed store
    const int kvh = hidx - 40;
    float x = base[NH*HD + NKV*HD + kvh*HD + d];
    vT[((size_t)kvh*HD + d) * S_LEN + s] = f32_to_bf16(x);
    return;
  }
  const float* src = (hidx < NH) ? (base + hidx*HD)
                                 : (base + NH*HD + (hidx-NH)*HD);
  float x = src[d];
  float ss = x * x;
#pragma unroll
  for (int m = 16; m >= 1; m >>= 1) ss += __shfl_xor(ss, m, 32);
  __shared__ float red[4];
  if ((threadIdx.x & 31) == 0) red[threadIdx.x >> 5] = ss;
  __syncthreads();
  float inv = rsqrtf((red[0]+red[1]+red[2]+red[3]) * (1.0f/HD) + 1e-6f);
  const float* w = (hidx < NH) ? qw : kw;
  float xn = w[d] * (x * inv);
  const int p = (d < 64) ? d + 64 : d - 64;
  float on = w[p] * (src[p] * inv);
  const int j = (d < 64) ? d : d - 64;
  float c  = cosb[(size_t)s*64 + j];
  float sn = sinb[(size_t)s*64 + j];
  float rot = (d < 64) ? (xn*c - on*sn) : (xn*c + on*sn);
  if (hidx < NH)
    qb[((size_t)hidx*S_LEN + s)*HD + d] = f32_to_bf16(rot);
  else
    kb[((size_t)(hidx-NH)*S_LEN + s)*HD + d] = f32_to_bf16(rot);
}

// -------------------------------------------------------------------------
// Flash-style causal attention with GQA (n_rep=4).
// grid (S/128, NH), block 256 (8 waves). Wave w owns 16 queries; loops over
// 128-key blocks (causally skipped), online fp32 softmax, bf16 WMMA for
// Q@K^T and P@V. All 8 B-fragments of a K-chunk are loaded as one clause
// before the 8 WMMAs. P re-laid-out C->A via a per-wave LDS tile (DS ops
// are in-order within a wave -> no barrier needed).
// -------------------------------------------------------------------------
__global__ __launch_bounds__(256) void attn_kernel(const u16* __restrict__ qb,
                                                   const u16* __restrict__ kbuf,
                                                   const u16* __restrict__ vT,
                                                   u16* __restrict__ attnb) {
  __shared__ u16 pS[8][16][136];   // per-wave 16x128 bf16 P tile (+pad)
  const int lane = threadIdx.x & 31;
  const int wave = threadIdx.x >> 5;
  const int la = lane & 15;
  const int hi = lane >> 4;
  const int h  = blockIdx.y;
  const int kv = h >> 2;                       // GQA: head h uses kv head h/4
  const int q0 = blockIdx.x * 128 + wave * 16;

  const u16* qbase = qb   + ((size_t)h * S_LEN + q0) * HD;
  const u16* kbase = kbuf + (size_t)kv * S_LEN * HD;
  const u16* vbase = vT   + (size_t)kv * HD * S_LEN;

  v8f o[8];
#pragma unroll
  for (int f = 0; f < 8; ++f) o[f] = v8f_zero();
  float m_run[8], l_run[8];
#pragma unroll
  for (int r = 0; r < 8; ++r) { m_run[r] = -3.0e38f; l_run[r] = 0.0f; }

  const int aoff = hi ? 16 : 0;
  // Q A-fragments over HD=128 (4 chunks of K=32), loaded once.
  v16bf qf[4];
#pragma unroll
  for (int kk = 0; kk < 4; ++kk)
    qf[kk] = load_afrag(qbase + (size_t)la*HD + aoff, kk*32);

  const int nkb = (q0 + 16 + 127) >> 7;        // causal block skipping
  for (int b = 0; b < nkb; ++b) {
    const int kb0 = b << 7;
    // ---- scores: 16 x 128 tile, 8 N-fragments ----
    v8f sc[8];
#pragma unroll
    for (int t = 0; t < 8; ++t) sc[t] = v8f_zero();
#pragma unroll
    for (int kk = 0; kk < 4; ++kk) {
      v16bf kf[8];
#pragma unroll
      for (int t = 0; t < 8; ++t)
        kf[t] = *reinterpret_cast<const v16bf*>(
            kbase + (size_t)(kb0 + t*16 + la) * HD + kk*32 + hi*16);
#pragma unroll
      for (int t = 0; t < 8; ++t)
        sc[t] = __builtin_amdgcn_wmma_f32_16x16x32_bf16(false, qf[kk], false, kf[t],
                                                        (short)0, sc[t], false, false);
    }
    // ---- scale + causal mask + running row max ----
    float mnew[8];
#pragma unroll
    for (int r = 0; r < 8; ++r) mnew[r] = m_run[r];
#pragma unroll
    for (int t = 0; t < 8; ++t) {
      const int key = kb0 + t*16 + la;
#pragma unroll
      for (int r = 0; r < 8; ++r) {
        const int qq = q0 + r + hi*8;
        float v = sc[t][r] * ATTN_SCALE;
        v = (key <= qq) ? v : -3.0e38f;
        sc[t][r] = v;
        mnew[r] = fmaxf(mnew[r], v);
      }
    }
#pragma unroll
    for (int r = 0; r < 8; ++r)
#pragma unroll
      for (int m = 1; m <= 8; m <<= 1)
        mnew[r] = fmaxf(mnew[r], __shfl_xor(mnew[r], m, 32));
    // ---- rescale accumulators ----
#pragma unroll
    for (int r = 0; r < 8; ++r) {
      float alpha = __expf(m_run[r] - mnew[r]);
      l_run[r] *= alpha;
      m_run[r] = mnew[r];
#pragma unroll
      for (int f = 0; f < 8; ++f) o[f][r] *= alpha;
    }
    // ---- exp, row sums, stage P (bf16) in LDS in (row, col) order ----
    float rs[8];
#pragma unroll
    for (int r = 0; r < 8; ++r) rs[r] = 0.0f;
#pragma unroll
    for (int t = 0; t < 8; ++t) {
#pragma unroll
      for (int r = 0; r < 8; ++r) {
        float p = __expf(sc[t][r] - m_run[r]);
        rs[r] += p;
        pS[wave][r + hi*8][t*16 + la] = f32_to_bf16(p);
      }
    }
#pragma unroll
    for (int r = 0; r < 8; ++r) {
#pragma unroll
      for (int m = 1; m <= 8; m <<= 1) rs[r] += __shfl_xor(rs[r], m, 32);
      l_run[r] += rs[r];
    }
    // ---- P @ V (contraction over 128 keys, 4 chunks of 32) ----
#pragma unroll
    for (int kk = 0; kk < 4; ++kk) {
      v16bf pf = load_afrag(&pS[wave][la][0] + aoff, kk*32);
      v16bf vf[8];
#pragma unroll
      for (int f = 0; f < 8; ++f)
        vf[f] = *reinterpret_cast<const v16bf*>(
            vbase + (size_t)(f*16 + la) * S_LEN + kb0 + kk*32 + hi*16);
#pragma unroll
      for (int f = 0; f < 8; ++f)
        o[f] = __builtin_amdgcn_wmma_f32_16x16x32_bf16(false, pf, false, vf[f],
                                                       (short)0, o[f], false, false);
    }
  }
  // ---- normalize and store attn (bf16, token-major [S, H*HD]) ----
#pragma unroll
  for (int r = 0; r < 8; ++r) {
    float invl = 1.0f / l_run[r];
    const int qq = q0 + r + hi*8;
#pragma unroll
    for (int f = 0; f < 8; ++f)
      attnb[(size_t)qq * (NH*HD) + h*HD + f*16 + la] = f32_to_bf16(o[f][r] * invl);
  }
}

// -------------------------------------------------------------------------
extern "C" void kernel_launch(void* const* d_in, const int* in_sizes, int n_in,
                              void* d_out, int out_size, void* d_ws, size_t ws_size,
                              hipStream_t stream) {
  (void)in_sizes; (void)n_in; (void)out_size; (void)ws_size;
  const float* hidden = (const float*)d_in[0];
  const float* cosb   = (const float*)d_in[1];
  const float* sinb   = (const float*)d_in[2];
  /* d_in[3] = attention_mask: pure causal; applied analytically in-kernel */
  const float* Wq  = (const float*)d_in[4];
  const float* Wk  = (const float*)d_in[5];
  const float* Wv  = (const float*)d_in[6];
  const float* Wo  = (const float*)d_in[7];
  const float* qnw = (const float*)d_in[8];
  const float* knw = (const float*)d_in[9];

  char* ws = (char*)d_ws;
  u16*   hb    = (u16*)(ws);                         // bf16 hidden   16 MB
  u16*   wqkv  = (u16*)(ws + ((size_t)16  << 20));   // bf16 Wq|Wk|Wv 48 MB
  u16*   wob   = (u16*)(ws + ((size_t)64  << 20));   // bf16 Wo       32 MB
  float* qkv   = (float*)(ws + ((size_t)96 << 20));  // f32 QKV       48 MB
  u16*   qb    = (u16*)(ws + ((size_t)144 << 20));   // bf16 q packed 16 MB
  u16*   kbuf  = (u16*)(ws + ((size_t)160 << 20));   // bf16 k packed  4 MB
  u16*   vT    = (u16*)(ws + ((size_t)164 << 20));   // bf16 v^T       4 MB
  u16*   attnb = (u16*)(ws + ((size_t)168 << 20));   // bf16 attn     16 MB
  float* out   = (float*)d_out;

  // 1) convert to bf16 (weights concatenated along N for one fused QKV GEMM)
  cvt_bf16_kernel<<<(S_LEN*DM)/256, 256, 0, stream>>>(hidden, hb, S_LEN*DM);
  cvt_bf16_kernel<<<(NH*HD*DM)/256, 256, 0, stream>>>(Wq, wqkv, NH*HD*DM);
  cvt_bf16_kernel<<<(NKV*HD*DM)/256, 256, 0, stream>>>(Wk, wqkv + (size_t)NH*HD*DM, NKV*HD*DM);
  cvt_bf16_kernel<<<(NKV*HD*DM)/256, 256, 0, stream>>>(Wv, wqkv + (size_t)(NH*HD + NKV*HD)*DM, NKV*HD*DM);
  cvt_bf16_kernel<<<(DM*NH*HD)/256, 256, 0, stream>>>(Wo, wob, DM*NH*HD);

  // 2) fused QKV projection: [2048,4096] x [6144,4096]^T -> [2048,6144] f32
  gemm_bf16_wmma<<<dim3(NQKV/256, S_LEN/64), 256, 0, stream>>>(hb, wqkv, qkv,
                                                               S_LEN, NQKV, DM);
  // 3) RMSNorm + RoPE + pack (q/k) and transpose-pack (v)
  normropepack_kernel<<<dim3(S_LEN, 48), 128, 0, stream>>>(qkv, cosb, sinb,
                                                           qnw, knw, qb, kbuf, vT);
  // 4) flash attention (causal, GQA)
  attn_kernel<<<dim3(S_LEN/128, NH), 256, 0, stream>>>(qb, kbuf, vT, attnb);
  // 5) output projection: [2048,4096] x [4096,4096]^T -> f32 d_out
  gemm_bf16_wmma<<<dim3(DM/256, S_LEN/64), 256, 0, stream>>>(attnb, wob, out,
                                                             S_LEN, DM, DM);
}
